// CliffordEPBottleneck_43241730736881
// MI455X (gfx1250) — matrix operand order repaired and checked by reading, not compiled, not run on hardware
//
#include <hip/hip_runtime.h>
#include <math.h>

// ---------------------------------------------------------------------------
// CliffordEPBottleneck on MI455X (gfx1250, wave32, WMMA + async-LDS copies).
//
// Per step (h stored as H[j][m], j = b*COMP + c, 1024 x 4096 f32):
//   rho = tanh(H)                              (bf16 copy Rb for the GEMM)
//   D[j][n] = sum_m Rb[j][m] * Wsym[n][m]      (bf16 WMMA, f32 accum)
//   Hn[j][n] = (1-DT)*H[j][n] + DT*(1-rho^2)*(D[j][n] + X[j][n])
// Wsym is symmetric -> its row-major rows serve directly as the WMMA
// B-operand; no transpose anywhere. Global->LDS staging uses CDNA5
// GLOBAL_LOAD_ASYNC_TO_LDS_B128 with double-buffered LDS (ASYNCcnt pipelined
// one K-chunk ahead); last chunk peeled so the steady-state loop body is
// branch-free.
// ---------------------------------------------------------------------------

#define DT_C    0.1f
#define NODES   4096
#define JDIM    1024          // B * COMP = 256 * 4
#define NFREE   10

typedef __bf16 bf16;
typedef __attribute__((ext_vector_type(4)))  bf16  v4bf;
typedef __attribute__((ext_vector_type(8)))  bf16  v8bf;
typedef __attribute__((ext_vector_type(16))) bf16  v16bf;
typedef __attribute__((ext_vector_type(4)))  float v4f;
typedef __attribute__((ext_vector_type(8)))  float v8f;

// Branch-free tanh: e = exp(-2|x|) in (0,1] -> no overflow; straight-line VALU
__device__ __forceinline__ float fast_tanh(float x) {
    float ax = __builtin_fabsf(x);
    float e  = __expf(-2.0f * ax);
    float t  = (1.0f - e) / (1.0f + e);
    return __builtin_copysignf(t, x);
}

// CDNA5 async memory->LDS copy, 16 bytes per lane (GVS addressing mode).
// vdst = per-lane LDS byte address, voff = per-lane 32-bit byte offset,
// sbase = uniform 64-bit base in SGPRs. Tracked with ASYNCcnt.
__device__ __forceinline__ void async_copy_b128(unsigned lds_addr, unsigned voff,
                                                const void* sbase) {
    asm volatile("global_load_async_to_lds_b128 %0, %1, %2"
                 :: "v"(lds_addr), "v"(voff), "s"(sbase)
                 : "memory");
}

// ---------------------------------------------------------------------------
// One-time prep: Wsym in bf16 (row-major [n][m], symmetric by construction)
// ---------------------------------------------------------------------------
__global__ void k_wsym(const float* __restrict__ W, bf16* __restrict__ Wb) {
    int i = blockIdx.x * blockDim.x + threadIdx.x;   // n*4096 + m
    int n = i >> 12;
    int m = i & (NODES - 1);
    float v = 0.5f * (W[i] + W[m * NODES + n]);
    Wb[i] = (bf16)v;
}

// One-time prep: X[j][m] = x[b, m*4+c] with j=b*4+c; H0 = X (h starts at x_mv)
__global__ void k_prepx(const float* __restrict__ x,
                        float* __restrict__ Xt, float* __restrict__ H0) {
    int i = blockIdx.x * blockDim.x + threadIdx.x;   // j*4096 + m
    int j = i >> 12;
    int m = i & (NODES - 1);
    int b = j >> 2, c = j & 3;
    float v = x[b * 16384 + m * 4 + c];
    Xt[i] = v;
    H0[i] = v;
}

// Per step: Rb = bf16(tanh(H)), 4-wide
__global__ void k_rho(const float* __restrict__ H, bf16* __restrict__ R) {
    int i = (blockIdx.x * blockDim.x + threadIdx.x) * 4;
    v4f h = *(const v4f*)(H + i);
    v4bf r;
    r.x = (bf16)fast_tanh(h.x);
    r.y = (bf16)fast_tanh(h.y);
    r.z = (bf16)fast_tanh(h.z);
    r.w = (bf16)fast_tanh(h.w);
    *(v4bf*)(R + i) = r;
}

// ---------------------------------------------------------------------------
// GEMM + fused Euler update.
//   C[1024 x 4096] = A(Rb, 1024x4096 bf16) * B(Wsym rows, bf16), K = 4096
// Tile: 128(j) x 128(n) per 256-thread block (8 waves: 2 j-waves x 4 n-waves,
// each wave 64x32 = 4x2 WMMA accumulators). BK = 32, double-buffered LDS fed
// by async-to-LDS copies pipelined one chunk ahead.
// ---------------------------------------------------------------------------
#define BM 128
#define BN 128
#define BK 32
#define LDT 40   // padded LDS row stride in bf16 (80 B; 16B-aligned chunks)

__launch_bounds__(256)
__global__ void k_gemm_update(const bf16*  __restrict__ A,    // Rb  [1024][4096]
                              const bf16*  __restrict__ Bw,   // Wb  [4096][4096]
                              const float* __restrict__ Hc,   // H   [1024][4096]
                              const float* __restrict__ Xt,   // X   [1024][4096]
                              float*       __restrict__ Hn) { // out [1024][4096]
    __shared__ __align__(16) bf16 sA[2][BM * LDT];
    __shared__ __align__(16) bf16 sB[2][BN * LDT];

    const int t    = threadIdx.x;
    const int lane = t & 31;
    const int wid  = t >> 5;
    const int wm   = wid >> 2;       // 0..1  (j direction, 64 rows each)
    const int wn   = wid & 3;        // 0..3  (n direction, 32 cols each)
    const int l16  = lane & 15;
    const int kh   = lane >> 4;      // lane half
    const int jb   = blockIdx.y * BM;
    const int nb   = blockIdx.x * BN;

    // This thread's two (row, chunk) stage slots: 512 b128 copies / operand
    const int idx0 = t,       row0 = idx0 >> 2, ch0 = (idx0 & 3) * 8;
    const int idx1 = t + 256, row1 = idx1 >> 2, ch1 = (idx1 & 3) * 8;

    v8f acc[4][2] = {};

    // Issue one K-chunk's async copies (4 per thread: 2 A-slots + 2 B-slots)
    auto issue = [&](int buf, int k0) {
        async_copy_b128((unsigned)(size_t)&sA[buf][row0 * LDT + ch0],
                        (unsigned)(((jb + row0) * NODES + k0 + ch0) * 2), A);
        async_copy_b128((unsigned)(size_t)&sB[buf][row0 * LDT + ch0],
                        (unsigned)(((nb + row0) * NODES + k0 + ch0) * 2), Bw);
        async_copy_b128((unsigned)(size_t)&sA[buf][row1 * LDT + ch1],
                        (unsigned)(((jb + row1) * NODES + k0 + ch1) * 2), A);
        async_copy_b128((unsigned)(size_t)&sB[buf][row1 * LDT + ch1],
                        (unsigned)(((nb + row1) * NODES + k0 + ch1) * 2), Bw);
    };

    // Fragment loads (per ISA VGPR layouts) + 8 WMMAs for one staged chunk
    auto compute = [&](int buf) {
        v16bf af[4], bfr[2];
#pragma unroll
        for (int mt = 0; mt < 4; ++mt) {
            // A 16x32: lane holds row M=l16; K = kh*8+(0..7) and 16+kh*8+(0..7)
            const bf16* base = &sA[buf][(wm * 64 + mt * 16 + l16) * LDT];
            v8bf lo = *(const v8bf*)(base + kh * 8);
            v8bf hi = *(const v8bf*)(base + 16 + kh * 8);
            af[mt] = __builtin_shufflevector(lo, hi,
                0,1,2,3,4,5,6,7,8,9,10,11,12,13,14,15);
        }
#pragma unroll
        for (int nt = 0; nt < 2; ++nt) {
            // B 32x16: lane holds col N=l16; K = kh*8+(0..7) and 16+kh*8+(0..7)
            const bf16* base = &sB[buf][(wn * 32 + nt * 16 + l16) * LDT + kh * 8];
            v8bf lo = *(const v8bf*)(base);
            v8bf hi = *(const v8bf*)(base + 16);
            bfr[nt] = __builtin_shufflevector(lo, hi,
                0,1,2,3,4,5,6,7,8,9,10,11,12,13,14,15);
        }
#pragma unroll
        for (int mt = 0; mt < 4; ++mt)
#pragma unroll
            for (int nt = 0; nt < 2; ++nt)
                acc[mt][nt] = __builtin_amdgcn_wmma_f32_16x16x32_bf16(
                    false, af[mt], false, bfr[nt],
                    (short)0, acc[mt][nt], false, false);
    };

    issue(0, 0);   // prologue

    // Steady state: branch-free body; last chunk peeled below
    int k0 = 0;
    for (; k0 < NODES - BK; k0 += BK) {
        const int buf = (k0 >> 5) & 1;
        issue(buf ^ 1, k0 + BK);   // prefetch next chunk into other buffer
        // async loads complete in order: oldest 4 (current chunk) done at <=4
        asm volatile("s_wait_asynccnt 0x4" ::: "memory");
        __syncthreads();
        compute(buf);
        __syncthreads();           // reads done before next issue overwrites
    }

    // Peeled final chunk
    asm volatile("s_wait_asynccnt 0x0" ::: "memory");
    __syncthreads();
    compute((k0 >> 5) & 1);

    // ---- fused epilogue: Euler step with input coupling
    // D layout: lane l16 = column N; vgpr r holds row M = r + 8*kh
#pragma unroll
    for (int mt = 0; mt < 4; ++mt) {
#pragma unroll
        for (int nt = 0; nt < 2; ++nt) {
            int n = nb + wn * 32 + nt * 16 + l16;
#pragma unroll
            for (int r = 0; r < 8; ++r) {
                int j = jb + wm * 64 + mt * 16 + kh * 8 + r;
                size_t off = (size_t)j * NODES + n;
                float h   = Hc[off];
                float rho = fast_tanh(h);
                float drive = acc[mt][nt][r] + Xt[off];
                Hn[off] = (1.0f - DT_C) * h
                        + DT_C * (1.0f - rho * rho) * drive;
            }
        }
    }
}

// Final gather: out[b][n] = H[j = b*4 + 0][n]   (grade-0 component)
__global__ void k_out(const float* __restrict__ H, float* __restrict__ out) {
    int i = blockIdx.x * blockDim.x + threadIdx.x;  // b*4096 + n
    int b = i >> 12;
    int n = i & (NODES - 1);
    out[i] = H[(size_t)(b * 4) * NODES + n];
}

// ---------------------------------------------------------------------------
extern "C" void kernel_launch(void* const* d_in, const int* in_sizes, int n_in,
                              void* d_out, int out_size, void* d_ws, size_t ws_size,
                              hipStream_t stream) {
    const float* x = (const float*)d_in[0];   // (256, 16384) f32
    const float* W = (const float*)d_in[1];   // (4096, 4096) f32

    // Workspace map (88 MiB total)
    char* ws = (char*)d_ws;
    bf16*  Wb = (bf16*) (ws);                                   // 32 MiB
    float* Xt = (float*)(ws + (size_t)32 * 1024 * 1024);        // 16 MiB
    float* Ha = (float*)(ws + (size_t)48 * 1024 * 1024);        // 16 MiB
    float* Hb = (float*)(ws + (size_t)64 * 1024 * 1024);        // 16 MiB
    bf16*  Rb = (bf16*) (ws + (size_t)80 * 1024 * 1024);        //  8 MiB

    // One-time prep (repeated per call; deterministic)
    k_wsym <<<(NODES * NODES) / 256, 256, 0, stream>>>(W, Wb);
    k_prepx<<<(JDIM * NODES) / 256, 256, 0, stream>>>(x, Xt, Ha);

    float* H[2] = {Ha, Hb};
    int cur = 0;
    dim3 ggrid(NODES / BN, JDIM / BM);   // (32, 8)
    for (int it = 0; it < NFREE; ++it) {
        k_rho<<<(JDIM * NODES) / (256 * 4), 256, 0, stream>>>(H[cur], Rb);
        k_gemm_update<<<ggrid, 256, 0, stream>>>(Rb, Wb, H[cur], Xt, H[cur ^ 1]);
        cur ^= 1;
    }
    k_out<<<(256 * NODES) / 256, 256, 0, stream>>>(H[cur], (float*)d_out);
}
